// NetworkEncoderRoPE_85968065396755
// MI455X (gfx1250) — compile-verified, matrix operand order checked by reading, not run on hardware
//
#include <hip/hip_runtime.h>
#include <hip/hip_bf16.h>
#include <math.h>

typedef __attribute__((ext_vector_type(16))) _Float16 v16h;
typedef __attribute__((ext_vector_type(8)))  _Float16 v8h;
typedef __attribute__((ext_vector_type(8)))  float    v8f;
typedef __attribute__((ext_vector_type(4)))  unsigned int u32x4;
typedef __attribute__((ext_vector_type(8)))  int i32x8;
typedef __attribute__((ext_vector_type(4)))  int i32x4;

#define DEV __device__ __forceinline__

// ---------------------------------------------------------------------------
// WMMA helpers (CDNA5 wave32, V_WMMA_F32_16X16X32_F16)
// Per-lane fragment = two contiguous 8-f16 chunks:
//   A frag: lane l -> row m = mBase + (l&15); chunks at k = kBase + ((l>>4)<<3)
//           and k+16.  B frag identical with n = nBase + (l&15) on W^T storage.
//   C/D: lane l, vgpr r -> (m = r + 8*(l>>4), n = l&15)
// ---------------------------------------------------------------------------
DEV v8f wmma_f16(v16h a, v16h b, v8f c) {
  return __builtin_amdgcn_wmma_f32_16x16x32_f16(false, a, false, b, (short)0, c,
                                                false, false);
}

DEV v16h frag16(const _Float16* p) {  // p must be 16B aligned
  v8h lo = *(const v8h*)p;
  v8h hi = *(const v8h*)(p + 16);
  return __builtin_shufflevector(lo, hi, 0, 1, 2, 3, 4, 5, 6, 7, 8, 9, 10, 11,
                                 12, 13, 14, 15);
}

DEV v8f v8f_zero() {
  v8f z;
#pragma unroll
  for (int r = 0; r < 8; ++r) z[r] = 0.f;
  return z;
}

// ---------------------------------------------------------------------------
// TDM: async DMA of a 128x32 f16 tile (row stride = rowStrideElems) into LDS
// with row padding 16 DWORDs -> +4 DWORDs (LDS row stride 40 f16).
// Descriptor per CDNA5 ISA ch.8 (D# group0/group1; groups 2-3 zero for 2D).
// ---------------------------------------------------------------------------
DEV void tdm_load_tile_a(const _Float16* gsrc, unsigned ldsOff,
                         unsigned rowStrideElems) {
  const unsigned long ga = (unsigned long)gsrc;
  u32x4 g0 = {0u, 0u, 0u, 0u};
  g0[0] = 1u;                                            // count=1, user D#
  g0[1] = ldsOff;                                        // lds_addr (bytes)
  g0[2] = (unsigned)ga;                                  // global_addr lo
  g0[3] = (unsigned)((ga >> 32) & 0x01ffffffu) | (2u << 30);  // addr hi | type=2
  i32x8 g1 = {0, 0, 0, 0, 0, 0, 0, 0};
  const unsigned dim0 = rowStrideElems;                  // tensor_dim0 (elems)
  const unsigned dim1 = 0x100000u;                       // tensor_dim1 (rows)
  // data_size=2B | pad_enable | pad_interval=16dw (code3) | pad_amount=4dw (code3)
  g1[0] = (int)((1u << 16) | (1u << 20) | (3u << 22) | (3u << 25));
  g1[1] = (int)((dim0 & 0xffffu) << 16);                 // tensor_dim0 lo16
  g1[2] = (int)((dim0 >> 16) | ((dim1 & 0xffffu) << 16));
  g1[3] = (int)((dim1 >> 16) | (32u << 16));             // tile_dim0 = 32
  g1[4] = (int)128;                                      // tile_dim1 = 128
  g1[5] = (int)dim0;                                     // tensor_dim0_stride
  i32x4 z4 = {0, 0, 0, 0};
#if __has_include(<hip/amd_detail/amd_gfx1250_TDM.h>)
  i32x8 z8 = {0, 0, 0, 0, 0, 0, 0, 0};
  __builtin_amdgcn_tensor_load_to_lds(g0, g1, z4, z4, z8, 0);
#else
  __builtin_amdgcn_tensor_load_to_lds(g0, g1, z4, z4, 0);
#endif
}

// ---------------------------------------------------------------------------
// Generic WMMA GEMM:  out = EP(A @ W + bias)
//   A: f16 [M x (nChunks*Kper)], up to 2 chunk pointers (row stride Kper)
//      -> A tiles streamed by the Tensor Data Mover, double-buffered in LDS
//   W: f32 [nChunks*Kper x N] row-major; staged into LDS transposed as f16
// Block tile 128x64, BK=32, 8 waves, each wave a 2x2 patch of 16x16 tiles.
// EP_F16P remaps rows m = b*120+l -> padded row b*128+l on store.
// ---------------------------------------------------------------------------
enum { EP_F16P = 0, EP_GELU = 1, EP_H = 2, EP_GATE = 3, EP_RESID = 4 };

template <int EP>
__global__ __launch_bounds__(256) void gemm_wmma(
    const _Float16* __restrict__ A0, const _Float16* __restrict__ A1,
    const float* __restrict__ W, const float* __restrict__ bias,
    int M, int N, int Kper, int nChunks,
    _Float16* __restrict__ out16, float* __restrict__ out32,
    const float* __restrict__ h32, const float* __restrict__ x32) {
  __shared__ alignas(16) _Float16 sA[2][128 * 40];  // TDM double buffer
  __shared__ alignas(16) _Float16 sBt[64 * 40];     // W^T tile: [n][k]

  const int tid = threadIdx.x;
  const int lane = tid & 31, wave = tid >> 5;
  const int waveM = wave >> 1, waveN = wave & 1;
  const int bM = blockIdx.y * 128, bN = blockIdx.x * 64;

  v8f acc[2][2];
#pragma unroll
  for (int i = 0; i < 2; ++i)
#pragma unroll
    for (int j = 0; j < 2; ++j) acc[i][j] = v8f_zero();

  const int sBn = tid & 63, sBk = (tid >> 6) * 8;  // 64 n x 32 k W tile
  const int half = lane >> 4, nn = lane & 15, kh = half << 3;
  const int nkC = Kper / 32, nkTotal = nChunks * nkC;

  if (wave == 0) {  // TDM: kick off tile 0
    const _Float16* At = A0 + (long)bM * Kper;
    tdm_load_tile_a(At, (unsigned)(unsigned long)&sA[0][0], (unsigned)Kper);
  }

  for (int t = 0; t < nkTotal; ++t) {
    const int buf = t & 1;
    const int ch = t / nkC, kt = t - ch * nkC, k0 = kt * 32;
    {  // stage W^T tile: coalesced column reads, one b128 LDS store
      const float* src = W + ((long)ch * Kper + k0 + sBk) * N + bN + sBn;
      v8h wv;
#pragma unroll
      for (int j = 0; j < 8; ++j) wv[j] = (_Float16)src[(long)j * N];
      *(v8h*)&sBt[sBn * 40 + sBk] = wv;
    }
    if (wave == 0) {
      if (t + 1 < nkTotal) {  // TDM: prefetch next A tile into other buffer
        const int ch1 = (t + 1) / nkC, kt1 = (t + 1) - ch1 * nkC;
        const _Float16* At = (ch1 ? A1 : A0) + (long)bM * Kper + kt1 * 32;
        tdm_load_tile_a(At, (unsigned)(unsigned long)&sA[buf ^ 1][0],
                        (unsigned)Kper);
        __builtin_amdgcn_s_wait_tensorcnt((short)1);  // tile t complete
      } else {
        __builtin_amdgcn_s_wait_tensorcnt((short)0);
      }
    }
    __syncthreads();
    if (t + 1 < nkTotal)  // gfx1250 global_prefetch of next W tile
      __builtin_prefetch(W + ((long)((t + 1) / nkC) * Kper +
                              ((t + 1) % nkC) * 32 + sBk) * N + bN + sBn, 0, 1);
    v16h a0 = frag16(&sA[buf][(waveM * 32 + nn) * 40 + kh]);
    v16h a1 = frag16(&sA[buf][(waveM * 32 + 16 + nn) * 40 + kh]);
    v16h b0 = frag16(&sBt[(waveN * 32 + nn) * 40 + kh]);
    v16h b1 = frag16(&sBt[(waveN * 32 + 16 + nn) * 40 + kh]);
    acc[0][0] = wmma_f16(a0, b0, acc[0][0]);
    acc[0][1] = wmma_f16(a0, b1, acc[0][1]);
    acc[1][0] = wmma_f16(a1, b0, acc[1][0]);
    acc[1][1] = wmma_f16(a1, b1, acc[1][1]);
    __syncthreads();
  }

#pragma unroll
  for (int i = 0; i < 2; ++i)
#pragma unroll
    for (int j = 0; j < 2; ++j) {
      const int tm = bM + waveM * 32 + i * 16 + half * 8;
      const int tn = bN + waveN * 32 + j * 16 + nn;
      const float bv = bias[tn];
#pragma unroll
      for (int r = 0; r < 8; ++r) {
        const int m = tm + r;
        const long idx = (long)m * N + tn;
        float val = acc[i][j][r] + bv;
        if (EP == EP_F16P) {
          const int mb = m / 120, ml = m - mb * 120;
          out16[((long)mb * 128 + ml) * N + tn] = (_Float16)val;
        } else if (EP == EP_GELU) {
          float g = 0.5f * val * (1.f + erff(val * 0.70710678f));
          out16[idx] = (_Float16)g;
        } else if (EP == EP_H) {
          out16[idx] = (_Float16)val;
          out32[idx] = val;
        } else if (EP == EP_GATE) {
          float g = 1.f / (1.f + __expf(-val));
          out32[idx] = x32[idx] + g * h32[idx];
        } else if (EP == EP_RESID) {
          out32[idx] = out32[idx] + val;
        }
      }
    }
}

// ---------------------------------------------------------------------------
// LayerNorm (D=512): one wave32 per row, f32 in -> f16 out
// ---------------------------------------------------------------------------
__global__ __launch_bounds__(256) void ln_f16(const float* __restrict__ x,
                                              const float* __restrict__ g,
                                              const float* __restrict__ b,
                                              _Float16* __restrict__ out, int rows) {
  const int row = blockIdx.x * 8 + (threadIdx.x >> 5);
  const int lane = threadIdx.x & 31;
  if (row >= rows) return;
  const float* xr = x + (long)row * 512;
  float vbuf[16];
#pragma unroll
  for (int q = 0; q < 4; ++q) {
    float4 p = ((const float4*)(xr + lane * 16))[q];
    vbuf[4 * q + 0] = p.x; vbuf[4 * q + 1] = p.y;
    vbuf[4 * q + 2] = p.z; vbuf[4 * q + 3] = p.w;
  }
  float s = 0.f, ss = 0.f;
#pragma unroll
  for (int j = 0; j < 16; ++j) { s += vbuf[j]; ss += vbuf[j] * vbuf[j]; }
#pragma unroll
  for (int m = 16; m >= 1; m >>= 1) {
    s += __shfl_xor(s, m, 32);
    ss += __shfl_xor(ss, m, 32);
  }
  const float mean = s * (1.f / 512.f);
  const float var = ss * (1.f / 512.f) - mean * mean;
  const float rstd = rsqrtf(var + 1e-5f);
  _Float16* o = out + (long)row * 512 + lane * 16;
#pragma unroll
  for (int j = 0; j < 16; ++j) {
    const int c = lane * 16 + j;
    o[j] = (_Float16)((vbuf[j] - mean) * rstd * g[c] + b[c]);
  }
}

__global__ void f32_to_f16(const float* __restrict__ x, _Float16* __restrict__ o,
                           long n) {
  long i = (long)blockIdx.x * blockDim.x + threadIdx.x;
  const long st = (long)gridDim.x * blockDim.x;
  for (; i < n; i += st) o[i] = (_Float16)x[i];
}

// time_adjacency (120x120 f32) -> zero-padded 128x128 f16
__global__ void pad_adj(const float* __restrict__ A, _Float16* __restrict__ o) {
  const int i = blockIdx.x * 256 + threadIdx.x;  // 64 blocks
  const int r = i >> 7, c = i & 127;
  o[i] = (r < 120 && c < 120) ? (_Float16)A[r * 120 + c] : (_Float16)0.f;
}

// zero rows 120..127 of a padded [256][128][cols] f16 buffer
__global__ void zero_pad_rows(_Float16* __restrict__ buf, int cols) {
  const long i = (long)blockIdx.x * 256 + threadIdx.x;
  const long b = i / (8 * cols);
  const long rem = i - b * (8 * cols);
  const long r = 120 + rem / cols, c = rem - (rem / cols) * cols;
  buf[(b * 128 + r) * cols + c] = (_Float16)0.f;
}

// ---------------------------------------------------------------------------
// sim + geo per batch (padded inputs [B][128][128], pad rows zero):
//   S1 = Ksim @ Qsim^T * DS^-0.5   (stored transposed in LDS)
//   geo = 0.5 * A_pad @ S1         (f32 [B][128][128])
// ---------------------------------------------------------------------------
__global__ __launch_bounds__(256) void simgeo(const _Float16* __restrict__ ksim,
                                              const _Float16* __restrict__ qsim,
                                              const _Float16* __restrict__ A16,
                                              float* __restrict__ geo) {
  __shared__ alignas(16) _Float16 sST[128 * 136];  // S1^T: [q][j]
  const int b = blockIdx.x;
  const int lane = threadIdx.x & 31, wave = threadIdx.x >> 5;
  const int mB = wave * 16;
  const int half = lane >> 4, nn = lane & 15, kh = half << 3;
  const _Float16* Kb = ksim + (long)b * 128 * 128;
  const _Float16* Qb = qsim + (long)b * 128 * 128;

  v8f acc[8];
#pragma unroll
  for (int j = 0; j < 8; ++j) acc[j] = v8f_zero();
  for (int kt = 0; kt < 4; ++kt) {
    v16h a = frag16(Kb + (long)(mB + nn) * 128 + kt * 32 + kh);
#pragma unroll
    for (int j = 0; j < 8; ++j) {
      v16h bt = frag16(Qb + (long)(j * 16 + nn) * 128 + kt * 32 + kh);  // B^T
      acc[j] = wmma_f16(a, bt, acc[j]);
    }
  }
  const float sc = 0.08838834764831845f;  // DS^-0.5
#pragma unroll
  for (int j = 0; j < 8; ++j)
#pragma unroll
    for (int r = 0; r < 8; ++r)  // store transposed: sST[q][jrow]
      sST[(j * 16 + nn) * 136 + mB + half * 8 + r] = (_Float16)(acc[j][r] * sc);
  __syncthreads();

#pragma unroll
  for (int j = 0; j < 8; ++j) acc[j] = v8f_zero();
  for (int kt = 0; kt < 4; ++kt) {
    v16h a = frag16(A16 + (long)(mB + nn) * 128 + kt * 32 + kh);
#pragma unroll
    for (int j = 0; j < 8; ++j) {
      v16h bb = frag16(&sST[(j * 16 + nn) * 136 + kt * 32 + kh]);
      acc[j] = wmma_f16(a, bb, acc[j]);
    }
  }
  float* gb = geo + (long)b * 128 * 128;
#pragma unroll
  for (int j = 0; j < 8; ++j)
#pragma unroll
    for (int r = 0; r < 8; ++r)
      gb[(mB + half * 8 + r) * 128 + j * 16 + nn] = 0.5f * acc[j][r];
}

// ---------------------------------------------------------------------------
// Attention per (b,h) on padded Q/K/V [B][128][512]:
//   S = QK^T/8 + geo (key mask), softmax in registers, O = P @ V.
//   K staged in LDS; V^T overlays the K buffer after scores; P f16 in LDS.
// ---------------------------------------------------------------------------
__global__ __launch_bounds__(256) void attn_kernel(
    const _Float16* __restrict__ Q, const _Float16* __restrict__ K,
    const _Float16* __restrict__ V, const float* __restrict__ geo,
    _Float16* __restrict__ O) {
  __shared__ alignas(16) _Float16 sK[128 * 72];    // K tile; later V^T [64][136]
  __shared__ alignas(16) _Float16 sP[128 * 136];   // probabilities f16
  _Float16* sVt = sK;
  const int h = blockIdx.x, b = blockIdx.y;
  const int tid = threadIdx.x, lane = tid & 31, wave = tid >> 5;
  const int hoff = h * 64;
  const long gbase = ((long)b * 128) * 512 + hoff;

  {  // stage K tile (128x64, pad rows already zero)
    const int row = tid >> 1, cb = (tid & 1) * 32;
    const uint4* s = (const uint4*)(K + gbase + (long)row * 512 + cb);
    uint4* d = (uint4*)&sK[row * 72 + cb];
    d[0] = s[0]; d[1] = s[1]; d[2] = s[2]; d[3] = s[3];
  }
  __syncthreads();

  const int mB = wave * 16;
  const int half = lane >> 4, nn = lane & 15, kh = half << 3;
  v8f s[8];
#pragma unroll
  for (int j = 0; j < 8; ++j) s[j] = v8f_zero();
  for (int kt = 0; kt < 2; ++kt) {
    v16h a = frag16(Q + gbase + (long)(mB + nn) * 512 + kt * 32 + kh);
#pragma unroll
    for (int j = 0; j < 8; ++j) {
      v16h bt = frag16(&sK[(j * 16 + nn) * 72 + kt * 32 + kh]);  // K^T
      s[j] = wmma_f16(a, bt, s[j]);
    }
  }
  // scale + geometry bias + key mask
  const float* gb = geo + (long)b * 128 * 128;
#pragma unroll
  for (int j = 0; j < 8; ++j)
#pragma unroll
    for (int r = 0; r < 8; ++r) {
      const int m = mB + half * 8 + r;
      const int n = j * 16 + nn;
      float v = s[j][r] * 0.125f + gb[m * 128 + n];
      s[j][r] = (n < 120) ? v : -1e30f;
    }
  // row softmax in registers (rows live in 16-lane half-groups)
  float mx[8], inv[8];
#pragma unroll
  for (int r = 0; r < 8; ++r) {
    float t = -1e30f;
#pragma unroll
    for (int j = 0; j < 8; ++j) t = fmaxf(t, s[j][r]);
#pragma unroll
    for (int d2 = 1; d2 < 16; d2 <<= 1) t = fmaxf(t, __shfl_xor(t, d2, 32));
    mx[r] = t;
  }
#pragma unroll
  for (int r = 0; r < 8; ++r) {
    float t = 0.f;
#pragma unroll
    for (int j = 0; j < 8; ++j) {
      float e = __expf(s[j][r] - mx[r]);
      s[j][r] = e;
      t += e;
    }
#pragma unroll
    for (int d2 = 1; d2 < 16; d2 <<= 1) t += __shfl_xor(t, d2, 32);
    inv[r] = 1.f / t;
  }
  __syncthreads();  // all waves done reading K before restaging V^T
#pragma unroll
  for (int j = 0; j < 8; ++j)
#pragma unroll
    for (int r = 0; r < 8; ++r)
      sP[(mB + half * 8 + r) * 136 + j * 16 + nn] = (_Float16)(s[j][r] * inv[r]);
  {  // stage V^T: coalesced strided reads, b128 LDS stores
    const int n = tid & 63, grp = tid >> 6;  // n = head dim, keys grp*32..+31
    const _Float16* vp = V + gbase + (long)grp * 32 * 512 + n;
    _Float16 tmp[32];
#pragma unroll
    for (int j = 0; j < 32; ++j) tmp[j] = vp[(long)j * 512];
    uint4* d = (uint4*)&sVt[n * 136 + grp * 32];
    const uint4* sq = (const uint4*)tmp;
    d[0] = sq[0]; d[1] = sq[1]; d[2] = sq[2]; d[3] = sq[3];
  }
  __syncthreads();

  v8f o[4];
#pragma unroll
  for (int j = 0; j < 4; ++j) o[j] = v8f_zero();
  for (int kt = 0; kt < 4; ++kt) {
    v16h a = frag16(&sP[(mB + nn) * 136 + kt * 32 + kh]);
#pragma unroll
    for (int j = 0; j < 4; ++j) {
      v16h bb = frag16(&sVt[(j * 16 + nn) * 136 + kt * 32 + kh]);
      o[j] = wmma_f16(a, bb, o[j]);
    }
  }
#pragma unroll
  for (int j = 0; j < 4; ++j)
#pragma unroll
    for (int r = 0; r < 8; ++r) {
      const int m = mB + half * 8 + r;
      if (m < 120)
        O[((long)b * 120 + m) * 512 + hoff + j * 16 + nn] = (_Float16)o[j][r];
    }
}

// ---------------------------------------------------------------------------
// Host orchestration
// ---------------------------------------------------------------------------
extern "C" void kernel_launch(void* const* d_in, const int* in_sizes, int n_in,
                              void* d_out, int out_size, void* d_ws, size_t ws_size,
                              hipStream_t stream) {
  (void)in_sizes; (void)n_in; (void)out_size; (void)ws_size;
  const float* x_pose = (const float*)d_in[0];
  const float* x_vis  = (const float*)d_in[1];
  const float* adj    = (const float*)d_in[2];
  const float* ln_q_g = (const float*)d_in[3];
  const float* ln_q_b = (const float*)d_in[4];
  const float* ln_kv_g = (const float*)d_in[5];
  const float* ln_kv_b = (const float*)d_in[6];
  const float* Wq = (const float*)d_in[7];   const float* bq = (const float*)d_in[8];
  const float* Wk = (const float*)d_in[9];   const float* bk = (const float*)d_in[10];
  const float* Wv = (const float*)d_in[11];  const float* bv = (const float*)d_in[12];
  const float* Wo = (const float*)d_in[13];  const float* bo = (const float*)d_in[14];
  const float* Wsq = (const float*)d_in[15]; const float* bsq = (const float*)d_in[16];
  const float* Wsk = (const float*)d_in[17]; const float* bsk = (const float*)d_in[18];
  const float* gW = (const float*)d_in[19];  const float* gb = (const float*)d_in[20];
  const float* fln_g = (const float*)d_in[21]; const float* fln_b = (const float*)d_in[22];
  const float* fW1 = (const float*)d_in[23]; const float* fb1 = (const float*)d_in[24];
  const float* fW2 = (const float*)d_in[25]; const float* fb2 = (const float*)d_in[26];

  const long BL = 256L * 120;   // 30720 rows (divisible by 128)
  const long nX = BL * 512;
  const long padX = 256L * 128 * 512;   // padded [B][128][512]
  const long padS = 256L * 128 * 128;   // padded [B][128][128]

  char* ws = (char*)d_ws;
  size_t off = 0;
  auto take = [&](size_t bytes) -> char* {
    off = (off + 255) & ~(size_t)255;
    char* p = ws + off;
    off += bytes;
    return p;
  };
  _Float16* xp16 = (_Float16*)take(nX * 2);
  _Float16* xv16 = (_Float16*)take(nX * 2);
  _Float16* A16  = (_Float16*)take(128 * 128 * 2);
  _Float16* qn16 = (_Float16*)take(nX * 2);
  _Float16* kn16 = (_Float16*)take(nX * 2);
  _Float16* Q16  = (_Float16*)take(padX * 2);  // padded; FFN hidden reuses span
  _Float16* K16  = (_Float16*)take(padX * 2);
  _Float16* V16  = (_Float16*)take(padX * 2);
  _Float16* qs16 = (_Float16*)take(padS * 2);
  _Float16* ks16 = (_Float16*)take(padS * 2);
  float* geo     = (float*)take(padS * 4);
  float* h32     = (float*)take(nX * 4);
  // buffer reuse (lifetimes disjoint):
  _Float16* ao16  = qn16;            // attention output (compact), after proj
  _Float16* h16   = kn16;            // Wo output f16, after projections
  _Float16* ln16  = (_Float16*)h32;  // FFN LN, after gate consumed h32
  _Float16* hid16 = Q16;             // FFN hidden (125.8MB) over Q..geo span

  f32_to_f16<<<2048, 256, 0, stream>>>(x_pose, xp16, nX);
  f32_to_f16<<<2048, 256, 0, stream>>>(x_vis, xv16, nX);
  pad_adj<<<64, 256, 0, stream>>>(adj, A16);

  const dim3 blk(256);
  const dim3 g512(512 / 64, BL / 128);
  const dim3 g128(128 / 64, BL / 128);
  const dim3 g2048(2048 / 64, BL / 128);

  for (int i = 0; i < 2; ++i) {
    const float* xq = i ? x_vis : x_pose;
    const float* xk = i ? x_pose : x_vis;
    const _Float16* xq16 = i ? xv16 : xp16;
    float* outp = (float*)d_out + (long)i * nX;

    // re-zero pad rows (Q16 span was clobbered by previous branch's FFN)
    zero_pad_rows<<<4096, blk, 0, stream>>>(Q16, 512);
    zero_pad_rows<<<4096, blk, 0, stream>>>(K16, 512);
    zero_pad_rows<<<4096, blk, 0, stream>>>(V16, 512);
    zero_pad_rows<<<1024, blk, 0, stream>>>(qs16, 128);
    zero_pad_rows<<<1024, blk, 0, stream>>>(ks16, 128);

    ln_f16<<<BL / 8, blk, 0, stream>>>(xq, ln_q_g + i * 512, ln_q_b + i * 512,
                                       qn16, (int)BL);
    ln_f16<<<BL / 8, blk, 0, stream>>>(xk, ln_kv_g + i * 512, ln_kv_b + i * 512,
                                       kn16, (int)BL);
    gemm_wmma<EP_F16P><<<g512, blk, 0, stream>>>(
        qn16, nullptr, Wq + (long)i * 512 * 512, bq + i * 512,
        (int)BL, 512, 512, 1, Q16, nullptr, nullptr, nullptr);
    gemm_wmma<EP_F16P><<<g512, blk, 0, stream>>>(
        kn16, nullptr, Wk + (long)i * 512 * 512, bk + i * 512,
        (int)BL, 512, 512, 1, K16, nullptr, nullptr, nullptr);
    gemm_wmma<EP_F16P><<<g512, blk, 0, stream>>>(
        kn16, nullptr, Wv + (long)i * 512 * 512, bv + i * 512,
        (int)BL, 512, 512, 1, V16, nullptr, nullptr, nullptr);
    gemm_wmma<EP_F16P><<<g128, blk, 0, stream>>>(
        qn16, nullptr, Wsq + (long)i * 512 * 128, bsq + i * 128,
        (int)BL, 128, 512, 1, qs16, nullptr, nullptr, nullptr);
    gemm_wmma<EP_F16P><<<g128, blk, 0, stream>>>(
        kn16, nullptr, Wsk + (long)i * 512 * 128, bsk + i * 128,
        (int)BL, 128, 512, 1, ks16, nullptr, nullptr, nullptr);

    simgeo<<<256, blk, 0, stream>>>(ks16, qs16, A16, geo);
    attn_kernel<<<dim3(8, 256), blk, 0, stream>>>(Q16, K16, V16, geo, ao16);

    gemm_wmma<EP_H><<<g512, blk, 0, stream>>>(
        ao16, nullptr, Wo + (long)i * 512 * 512, bo + i * 512,
        (int)BL, 512, 512, 1, h16, h32, nullptr, nullptr);
    // gate: sigmoid([x, h] @ gate_W + b); out = x + g * h  -> d_out
    gemm_wmma<EP_GATE><<<g512, blk, 0, stream>>>(
        xq16, h16, gW + (long)i * 1024 * 512, gb + i * 512,
        (int)BL, 512, 512, 2, nullptr, outp, h32, xq);
    // FFN with residual into d_out
    ln_f16<<<BL / 8, blk, 0, stream>>>(outp, fln_g + i * 512, fln_b + i * 512,
                                       ln16, (int)BL);
    gemm_wmma<EP_GELU><<<g2048, blk, 0, stream>>>(
        ln16, nullptr, fW1 + (long)i * 512 * 2048, fb1 + i * 2048,
        (int)BL, 2048, 512, 1, hid16, nullptr, nullptr, nullptr);
    gemm_wmma<EP_RESID><<<g512, blk, 0, stream>>>(
        hid16, nullptr, fW2 + (long)i * 2048 * 512, fb2 + i * 512,
        (int)BL, 512, 2048, 1, nullptr, outp, nullptr, nullptr);
  }
}
